// EncoderLayer_71004399337823
// MI455X (gfx1250) — matrix operand order repaired and checked by reading, not compiled, not run on hardware
//
#include <hip/hip_runtime.h>
#include <math.h>
#include <stdint.h>

typedef __bf16 bf16_t;
typedef __attribute__((ext_vector_type(16))) __bf16 v16bf;
typedef __attribute__((ext_vector_type(8)))  __bf16 v8bf;
typedef __attribute__((ext_vector_type(8)))  float  v8f;
typedef __attribute__((ext_vector_type(4)))  unsigned int u32x4;
typedef __attribute__((ext_vector_type(8)))  int i32x8;
typedef __attribute__((ext_vector_type(4)))  int i32x4;

#define DM    1024
#define FF    4096
#define SQ    2048
#define NB    2
#define NH    16
#define DEPTH 64
#define MT    (NB*SQ)   /* 4096 token rows */

__device__ __forceinline__ v8f wmma_bf16(v16bf a, v16bf b, v8f c) {
  return __builtin_amdgcn_wmma_f32_16x16x32_bf16(false, a, false, b, (short)0, c, false, false);
}
__device__ __forceinline__ v16bf join16(v8bf lo, v8bf hi) {
  return __builtin_shufflevector(lo, hi, 0,1,2,3,4,5,6,7,8,9,10,11,12,13,14,15);
}

#if __has_builtin(__builtin_amdgcn_tensor_load_to_lds)
#define USE_TDM 1
// Build a 2D Tensor-DMA descriptor (ISA ch.8) and issue TENSOR_LOAD_TO_LDS.
// bf16 elements (data_size=1 -> 2 bytes). Optional LDS row padding.
// This toolchain's builtin takes 6 args (g0, g1, g2, g3, extra group, cpol).
__device__ __forceinline__ void tdm_load_2d(unsigned lds_addr, const void* gaddr,
                                            unsigned tile_w, unsigned tile_h,
                                            unsigned tensor_w, unsigned tensor_h,
                                            unsigned row_stride,
                                            int pad_en, int pad_interval, int pad_amount)
{
  unsigned long long ga = (unsigned long long)(uintptr_t)gaddr;
  u32x4 g0;
  g0[0] = 1u;                                              // count=1 (user D#)
  g0[1] = lds_addr;                                        // lds_addr [63:32]
  g0[2] = (unsigned)(ga & 0xFFFFFFFFu);                    // global_addr[31:0]
  g0[3] = (unsigned)((ga >> 32) & 0x1FFFFFFu) | (2u << 30);// addr[56:32] | type=2
  i32x8 g1;
  unsigned d0 = (1u << 16);                                // data_size=1 (2 bytes)
  if (pad_en) d0 |= (1u << 20) | ((unsigned)pad_interval << 22) | ((unsigned)pad_amount << 25);
  g1[0] = (int)d0;                                         // wg_mask=0 (not in cluster)
  g1[1] = (int)((tensor_w & 0xFFFFu) << 16);               // tensor_dim0[15:0]
  g1[2] = (int)(((tensor_w >> 16) & 0xFFFFu) | ((tensor_h & 0xFFFFu) << 16));
  g1[3] = (int)(((tensor_h >> 16) & 0xFFFFu) | ((tile_w & 0xFFFFu) << 16)); // tile_dim0
  g1[4] = (int)(tile_h & 0xFFFFu);                         // tile_dim1, tile_dim2=0
  g1[5] = (int)row_stride;                                 // tensor_dim0_stride[31:0]
  g1[6] = 0;                                               // stride[47:32], dim1_stride lo
  g1[7] = 0;
  i32x4 gz = {0, 0, 0, 0};                                 // groups 2/3 unused (2D)
  i32x8 z8 = {0, 0, 0, 0, 0, 0, 0, 0};
  __builtin_amdgcn_tensor_load_to_lds(g0, g1, gz, gz, z8, 0);
}
#else
#define USE_TDM 0
#endif

// ---------------------------------------------------------------- fp32 -> bf16
__global__ void cvt_kernel(const float* __restrict__ in, bf16_t* __restrict__ out, size_t n) {
  for (size_t i = (size_t)blockIdx.x * 256 + threadIdx.x; i < n; i += (size_t)gridDim.x * 256)
    out[i] = (bf16_t)in[i];
}

// fp32 [K][N] -> bf16 transposed [N][K]
__global__ void cvtT_kernel(const float* __restrict__ in, bf16_t* __restrict__ out, int K, int N) {
  const size_t total = (size_t)K * N;
  for (size_t i = (size_t)blockIdx.x * 256 + threadIdx.x; i < total; i += (size_t)gridDim.x * 256) {
    size_t n = i / (size_t)K, k = i % (size_t)K;
    out[i] = (bf16_t)in[k * (size_t)N + n];
  }
}

// fragment MAC on one staged 64x64 tile pair
__device__ __forceinline__ void tile_mac(const bf16_t* Asb, const bf16_t* Bsb,
                                         int am, int nt, int lm, int lh,
                                         v8f& acc0, v8f& acc1)
{
#pragma unroll
  for (int kc = 0; kc < 2; ++kc) {
    const bf16_t* ap  = &Asb[am * 64 + kc * 32 + 8 * lh];              // k = 8h.. / 8h+16..
    v16bf af = join16(*(const v8bf*)ap, *(const v8bf*)(ap + 16));
    const bf16_t* bp0 = &Bsb[(nt * 32 + lm) * 72 + kc * 32 + 8 * lh];  // k = 8h.. / 8h+16..
    v16bf b0 = join16(*(const v8bf*)bp0, *(const v8bf*)(bp0 + 16));
    const bf16_t* bp1 = bp0 + 16 * 72;
    v16bf b1 = join16(*(const v8bf*)bp1, *(const v8bf*)(bp1 + 16));
    acc0 = wmma_bf16(af, b0, acc0);
    acc1 = wmma_bf16(af, b1, acc1);
  }
}

// ---------------------------------------------------------------- generic WMMA GEMM
// C(MxN,f32) = A(MxK,bf16 row-major) @ Bt(NxK,bf16 row-major == B^T) + bias
//              [+resid] [relu] -> outF / outB.  TDM double-buffered staging.
__global__ __launch_bounds__(256)
void gemm_bf16_kernel(const bf16_t* __restrict__ A, const bf16_t* __restrict__ Bt,
                      const float* __restrict__ bias, const float* __restrict__ resid,
                      float* __restrict__ outF, bf16_t* __restrict__ outB,
                      int M, int N, int K, int relu)
{
  __shared__ bf16_t As[2][64 * 64];     // [row][k]
  __shared__ bf16_t Bs[2][64 * 72];     // [col][k], stride 72 (TDM pad / manual)
  const int row0 = blockIdx.x * 64, col0 = blockIdx.y * 64;
  const int tid = threadIdx.x;
  const int wave = tid >> 5, lane = tid & 31, lm = lane & 15, lh = lane >> 4;
  const int mt = wave & 3, nt = wave >> 2;    // 4x2 waves -> 64x64 tile

  v8f acc0 = {}; v8f acc1 = {};
  const int am = mt * 16 + lm;

#if USE_TDM
  const int T = K / 64;
  if (wave == 0) {   // TDM is wave-level; one wave drives the DMA for the block
    tdm_load_2d((unsigned)(uintptr_t)&As[0][0], &A [(size_t)row0 * K], 64, 64,
                K, 1u << 20, K, 0, 0, 0);
    tdm_load_2d((unsigned)(uintptr_t)&Bs[0][0], &Bt[(size_t)col0 * K], 64, 64,
                K, 1u << 20, K, /*pad*/1, /*32 dw*/4, /*4 dw*/3);
  }
  for (int t = 0; t < T; ++t) {
    const int buf = t & 1;
    if (wave == 0) {
      if (t + 1 < T) {   // issue next tile, wait for current (in-order per wave)
        tdm_load_2d((unsigned)(uintptr_t)&As[buf ^ 1][0],
                    &A [(size_t)row0 * K + (size_t)(t + 1) * 64], 64, 64,
                    K, 1u << 20, K, 0, 0, 0);
        tdm_load_2d((unsigned)(uintptr_t)&Bs[buf ^ 1][0],
                    &Bt[(size_t)col0 * K + (size_t)(t + 1) * 64], 64, 64,
                    K, 1u << 20, K, 1, 4, 3);
        __builtin_amdgcn_s_wait_tensorcnt(2);
      } else {
        __builtin_amdgcn_s_wait_tensorcnt(0);
      }
    }
    __syncthreads();
    tile_mac(As[buf], Bs[buf], am, nt, lm, lh, acc0, acc1);
    __syncthreads();
  }
#else
  for (int k0 = 0; k0 < K; k0 += 64) {
    {
      int e0 = tid * 8;
      int r0 = e0 >> 6, c0 = e0 & 63;
      *(v8bf*)&As[0][e0]           = *(const v8bf*)&A [(size_t)(row0 + r0) * K + k0 + c0];
      *(v8bf*)&Bs[0][r0 * 72 + c0] = *(const v8bf*)&Bt[(size_t)(col0 + r0) * K + k0 + c0];
      int e1 = e0 + 2048;
      int r1 = e1 >> 6, c1 = e1 & 63;
      *(v8bf*)&As[0][e1]           = *(const v8bf*)&A [(size_t)(row0 + r1) * K + k0 + c1];
      *(v8bf*)&Bs[0][r1 * 72 + c1] = *(const v8bf*)&Bt[(size_t)(col0 + r1) * K + k0 + c1];
    }
    __syncthreads();
    tile_mac(As[0], Bs[0], am, nt, lm, lh, acc0, acc1);
    __syncthreads();
  }
#endif

  const int cn0 = col0 + nt * 32 + lm;
#pragma unroll
  for (int r = 0; r < 8; ++r) {               // D layout: M = r + 8*lh, N = lane&15
    const size_t rr = (size_t)(row0 + mt * 16 + 8 * lh + r);
    float v0 = acc0[r] + bias[cn0];
    float v1 = acc1[r] + bias[cn0 + 16];
    if (resid) { v0 += resid[rr * N + cn0]; v1 += resid[rr * N + cn0 + 16]; }
    if (relu)  { v0 = v0 > 0.f ? v0 : 0.f;  v1 = v1 > 0.f ? v1 : 0.f; }
    if (outF)  { outF[rr * N + cn0] = v0;   outF[rr * N + cn0 + 16] = v1; }
    if (outB)  { outB[rr * N + cn0] = (bf16_t)v0; outB[rr * N + cn0 + 16] = (bf16_t)v1; }
  }
}

// ---------------------------------------------------------------- attention
#define LG_OFF   0
#define QS_OFF   (32*2048*4)
#define VS_OFF   (QS_OFF + 32*64*2)
#define RED_OFF  (VS_OFF + 64*40*2)
#define RINV_OFF (RED_OFF + 32*8*4)
#define ATTN_SMEM (RINV_OFF + 32*4)

__global__ __launch_bounds__(256)
void attn_kernel(const bf16_t* __restrict__ Q, const bf16_t* __restrict__ Kc,
                 const bf16_t* __restrict__ V, const int* __restrict__ mask,
                 bf16_t* __restrict__ ctx)
{
  extern __shared__ char smem[];
  float*  lg   = (float*)(smem + LG_OFF);     // [32][2048] resident logits
  bf16_t* qs   = (bf16_t*)(smem + QS_OFF);    // [32][64]
  bf16_t* vst  = (bf16_t*)(smem + VS_OFF);    // [64 depth][40 pad] transposed V block
  float*  red  = (float*)(smem + RED_OFF);    // [32][8]
  float*  rinv = (float*)(smem + RINV_OFF);   // [32]

  const int tid = threadIdx.x, wave = tid >> 5, lane = tid & 31;
  const int lm = lane & 15, lh = lane >> 4;
  const int blk = blockIdx.x;
  const int qt = blk & 63, h = (blk >> 6) & 15, b = blk >> 10;
  const int q0 = qt * 32;
  const size_t base = ((size_t)b * SQ) * DM + (size_t)h * DEPTH;

#if USE_TDM
  if (wave == 0) {
    tdm_load_2d((unsigned)(uintptr_t)qs, &Q[base + (size_t)q0 * DM], 64, 32,
                64, 1u << 20, DM, 0, 0, 0);
    __builtin_amdgcn_s_wait_tensorcnt(0);
  }
#else
  {
    int e = tid * 8;
    int r = e >> 6, c = e & 63;
    *(v8bf*)&qs[e] = *(const v8bf*)&Q[base + (size_t)(q0 + r) * DM + c];
  }
#endif
  __syncthreads();

  // --- Phase A: logits = Q K^T / sqrt(d), masked, into LDS ---
  for (int kb = wave; kb < SQ / 16; kb += 8) {
    v8f a0 = {}; v8f a1 = {};
    const int key = kb * 16 + lm;
#pragma unroll
    for (int kc = 0; kc < 2; ++kc) {
      const bf16_t* kp = &Kc[base + (size_t)key * DM + kc * 32 + 8 * lh];
      v16bf bfK = join16(*(const v8bf*)kp, *(const v8bf*)(kp + 16));
      const bf16_t* qp0 = &qs[lm * 64 + kc * 32 + 8 * lh];
      v16bf af0 = join16(*(const v8bf*)qp0, *(const v8bf*)(qp0 + 16));
      const bf16_t* qp1 = qp0 + 16 * 64;
      v16bf af1 = join16(*(const v8bf*)qp1, *(const v8bf*)(qp1 + 16));
      a0 = wmma_bf16(af0, bfK, a0);
      a1 = wmma_bf16(af1, bfK, a1);
    }
    const int mk = mask[b * SQ + key];
#pragma unroll
    for (int r = 0; r < 8; ++r) {
      float s0 = (mk == 0) ? -1.0e9f : a0[r] * 0.125f;
      float s1 = (mk == 0) ? -1.0e9f : a1[r] * 0.125f;
      lg[(8 * lh + r) * SQ + key]      = s0;
      lg[(16 + 8 * lh + r) * SQ + key] = s1;
    }
  }
  __syncthreads();

  // --- Phase B: softmax rows in LDS ---
  {
    const int r = tid >> 3, sub = tid & 7;
    const int c0 = sub * 256;
    float mx = -3.0e38f;
    for (int c = c0; c < c0 + 256; ++c) mx = fmaxf(mx, lg[r * SQ + c]);
    red[r * 8 + sub] = mx;
    __syncthreads();
    if (sub == 0) { float m2 = red[r*8]; for (int i = 1; i < 8; ++i) m2 = fmaxf(m2, red[r*8+i]); red[r*8] = m2; }
    __syncthreads();
    const float rowmax = red[r * 8];
    __syncthreads();
    float sum = 0.f;
    for (int c = c0; c < c0 + 256; ++c) { float e = __expf(lg[r*SQ+c] - rowmax); lg[r*SQ+c] = e; sum += e; }
    red[r * 8 + sub] = sum;
    __syncthreads();
    if (sub == 0) { float s2 = 0.f; for (int i = 0; i < 8; ++i) s2 += red[r*8+i]; rinv[r] = 1.0f / s2; }
    __syncthreads();
  }

  // --- Phase C: ctx = P @ V, V blocks transposed through LDS ---
  const int qh = wave >> 2, nc = wave & 3;   // 2x4 waves -> 32 rows x 64 depth
  v8f acc = {};
  const int m = qh * 16 + lm;
  for (int kt = 0; kt < SQ / 32; ++kt) {
    {
      int e = tid * 8;                      // 2048 elems, one pass
      int r = e >> 6, c = e & 63;
      v8bf vv = *(const v8bf*)&V[base + (size_t)(kt * 32 + r) * DM + c];
#pragma unroll
      for (int j = 0; j < 8; ++j) vst[(c + j) * 40 + r] = vv[j];
    }
    __syncthreads();
    const float rs = rinv[m];
    const float* pp = &lg[m * SQ + kt * 32 + 8 * lh];
    v8f plo = *(const v8f*)pp;
    v8f phi = *(const v8f*)(pp + 16);
    v16bf af;
#pragma unroll
    for (int i = 0; i < 8; ++i) {
      af[i]     = (bf16_t)(plo[i] * rs);
      af[i + 8] = (bf16_t)(phi[i] * rs);
    }
    const bf16_t* vp = &vst[(nc * 16 + lm) * 40 + 8 * lh];
    v16bf bfV = join16(*(const v8bf*)vp, *(const v8bf*)(vp + 16));
    acc = wmma_bf16(af, bfV, acc);
    __syncthreads();
  }
#pragma unroll
  for (int r = 0; r < 8; ++r) {
    const int mm = qh * 16 + 8 * lh + r;
    ctx[base + (size_t)(q0 + mm) * DM + nc * 16 + lm] = (bf16_t)acc[r];
  }
}

// ---------------------------------------------------------------- layernorm (fp32)
__global__ __launch_bounds__(256)
void ln_kernel(const float* __restrict__ in, const float* __restrict__ g,
               const float* __restrict__ b, float* __restrict__ outF,
               bf16_t* __restrict__ outB)
{
  __shared__ float buf[256];
  const int row = blockIdx.x, tid = threadIdx.x;
  const float* x = in + (size_t)row * DM;
  float s = 0.f;
  for (int c = tid; c < DM; c += 256) s += x[c];
  buf[tid] = s; __syncthreads();
  for (int o = 128; o > 0; o >>= 1) { if (tid < o) buf[tid] += buf[tid + o]; __syncthreads(); }
  const float mean = buf[0] * (1.0f / DM);
  __syncthreads();
  float v = 0.f;
  for (int c = tid; c < DM; c += 256) { float d = x[c] - mean; v += d * d; }
  buf[tid] = v; __syncthreads();
  for (int o = 128; o > 0; o >>= 1) { if (tid < o) buf[tid] += buf[tid + o]; __syncthreads(); }
  const float inv = rsqrtf(buf[0] * (1.0f / DM) + 1e-5f);
  for (int c = tid; c < DM; c += 256) {
    float y = (x[c] - mean) * inv * g[c] + b[c];
    if (outF) outF[(size_t)row * DM + c] = y;
    if (outB) outB[(size_t)row * DM + c] = (bf16_t)y;
  }
}

// ---------------------------------------------------------------- host orchestration
extern "C" void kernel_launch(void* const* d_in, const int* in_sizes, int n_in,
                              void* d_out, int out_size, void* d_ws, size_t ws_size,
                              hipStream_t stream)
{
  (void)in_sizes; (void)n_in; (void)out_size; (void)ws_size;
  const float* x     = (const float*)d_in[0];
  const int*   mask  = (const int*)  d_in[1];
  const float* wq_w  = (const float*)d_in[2];
  const float* wq_b  = (const float*)d_in[3];
  const float* wk_w  = (const float*)d_in[4];
  const float* wk_b  = (const float*)d_in[5];
  const float* wv_w  = (const float*)d_in[6];
  const float* wv_b  = (const float*)d_in[7];
  const float* wo_w  = (const float*)d_in[8];
  const float* wo_b  = (const float*)d_in[9];
  const float* w1    = (const float*)d_in[10];
  const float* b1    = (const float*)d_in[11];
  const float* w2    = (const float*)d_in[12];
  const float* b2    = (const float*)d_in[13];
  const float* ln1_g = (const float*)d_in[14];
  const float* ln1_b = (const float*)d_in[15];
  const float* ln2_g = (const float*)d_in[16];
  const float* ln2_b = (const float*)d_in[17];
  float* out = (float*)d_out;

  char* ws = (char*)d_ws;
  size_t off = 0;
  auto alloc = [&](size_t bytes) { char* p = ws + off; off += (bytes + 255) & ~(size_t)255; return p; };

  bf16_t* xb   = (bf16_t*)alloc((size_t)MT * DM * 2);   // row-major activations
  bf16_t* wqt  = (bf16_t*)alloc((size_t)DM * DM * 2);   // transposed weights [N][K]
  bf16_t* wkt  = (bf16_t*)alloc((size_t)DM * DM * 2);
  bf16_t* wvt  = (bf16_t*)alloc((size_t)DM * DM * 2);
  bf16_t* wot  = (bf16_t*)alloc((size_t)DM * DM * 2);
  bf16_t* w1t  = (bf16_t*)alloc((size_t)DM * FF * 2);   // [FF][DM]
  bf16_t* w2t  = (bf16_t*)alloc((size_t)FF * DM * 2);   // [DM][FF]
  bf16_t* Qb   = (bf16_t*)alloc((size_t)MT * DM * 2);
  bf16_t* Kb   = (bf16_t*)alloc((size_t)MT * DM * 2);
  bf16_t* Vb   = (bf16_t*)alloc((size_t)MT * DM * 2);
  bf16_t* ctxb = (bf16_t*)alloc((size_t)MT * DM * 2);
  float*  sum1 = (float*) alloc((size_t)MT * DM * 4);
  float*  out1f= (float*) alloc((size_t)MT * DM * 4);
  bf16_t* out1b= (bf16_t*)alloc((size_t)MT * DM * 2);
  bf16_t* h1b  = (bf16_t*)alloc((size_t)MT * FF * 2);
  float*  sum2 = (float*) alloc((size_t)MT * DM * 4);

  cvt_kernel <<<1024, 256, 0, stream>>>(x, xb, (size_t)MT * DM);
  cvtT_kernel<<<1024, 256, 0, stream>>>(wq_w, wqt, DM, DM);
  cvtT_kernel<<<1024, 256, 0, stream>>>(wk_w, wkt, DM, DM);
  cvtT_kernel<<<1024, 256, 0, stream>>>(wv_w, wvt, DM, DM);
  cvtT_kernel<<<1024, 256, 0, stream>>>(wo_w, wot, DM, DM);
  cvtT_kernel<<<1024, 256, 0, stream>>>(w1,   w1t, DM, FF);
  cvtT_kernel<<<1024, 256, 0, stream>>>(w2,   w2t, FF, DM);

  dim3 gDD(MT / 64, DM / 64);
  dim3 gDF(MT / 64, FF / 64);

  gemm_bf16_kernel<<<gDD, 256, 0, stream>>>(xb, wqt, wq_b, nullptr, nullptr, Qb, MT, DM, DM, 0);
  gemm_bf16_kernel<<<gDD, 256, 0, stream>>>(xb, wkt, wk_b, nullptr, nullptr, Kb, MT, DM, DM, 0);
  gemm_bf16_kernel<<<gDD, 256, 0, stream>>>(xb, wvt, wv_b, nullptr, nullptr, Vb, MT, DM, DM, 0);

  attn_kernel<<<NB * NH * (SQ / 32), 256, ATTN_SMEM, stream>>>(Qb, Kb, Vb, mask, ctxb);

  gemm_bf16_kernel<<<gDD, 256, 0, stream>>>(ctxb, wot, wo_b, x, sum1, nullptr, MT, DM, DM, 0);
  ln_kernel<<<MT, 256, 0, stream>>>(sum1, ln1_g, ln1_b, out1f, out1b);

  gemm_bf16_kernel<<<gDF, 256, 0, stream>>>(out1b, w1t, b1, nullptr, nullptr, h1b, MT, FF, DM, 1);
  gemm_bf16_kernel<<<gDD, 256, 0, stream>>>(h1b, w2t, b2, out1f, sum2, nullptr, MT, DM, FF, 0);
  ln_kernel<<<MT, 256, 0, stream>>>(sum2, ln2_g, ln2_b, out, nullptr);
}
